// T_GRUA_85761906966769
// MI455X (gfx1250) — matrix operand order repaired and checked by reading, not compiled, not run on hardware
//
#include <hip/hip_runtime.h>

// ---------------- problem constants (from reference) ----------------
#define B_      256
#define NUM_ENT 100000
#define NEI     128
#define NUM_REL 5000
#define S_      32
#define D_      128
#define TOPK    64
#define NG      (3 * D_)          // 384 gate columns
#define MROWS   (B_ * TOPK)       // 16384 GRU rows per step
#define HB      (3 * TOPK * D_)   // batch stride (floats) inside tree_emb_all

// ---------------- d_out layout (floats, concatenated in return order) ----
#define N3K     (B_ * 3 * TOPK)           // 49152
#define OFF_TN  0
#define OFF_TE  (N3K)                     // tree_emb_all [B,3,K,D]
#define OFF_PI  (N3K + N3K * D_)
#define OFF_PN  (OFF_PI + N3K)
#define OFF_AR  (OFF_PN + N3K)

// ---------------- workspace layout (bytes) ----------------
#define WOFF_MAXED   0                      // 5000 f32
#define WOFF_RELEMB  32768                  // 5000*128 bf16 = 1,280,000 B
#define WOFF_WIH     (WOFF_RELEMB + 1280000) // 384*128 bf16
#define WOFF_WHH     (WOFF_WIH + 98304)
#define WOFF_INT     (WOFF_WHH + 98304)     // 9 int arrays of 16384 each
#define ISZ          (MROWS * 4)

typedef __attribute__((ext_vector_type(16))) __bf16        v16bf;
typedef __attribute__((ext_vector_type(8)))  float         v8f;
typedef __attribute__((ext_vector_type(4)))  unsigned int  uint4v;
typedef __attribute__((ext_vector_type(8)))  unsigned int  uint8v;

__device__ __forceinline__ unsigned short bf16_of(float f) {
    unsigned int u = __float_as_uint(f);
    u = (u + 0x7FFFu + ((u >> 16) & 1u)) >> 16;   // round-to-nearest-even
    return (unsigned short)u;
}
__device__ __forceinline__ float sigm(float x) {
    return 1.0f / (1.0f + __expf(-x));
}

// ---------------------------------------------------------------
// TDM: DMA a dense 2D bf16 matrix [dim1 rows x dim0 cols] from global
// into LDS at byte offset lds_off.  D# per CDNA5 ISA ch.8 (§8.3/8.4).
// Issued once per calling wave; tracked with TENSORcnt; EXEC ignored.
// ---------------------------------------------------------------
__device__ __forceinline__ void tdm_load_2d_bf16(
    const void* gptr, unsigned lds_off,
    unsigned dim0, unsigned dim1, unsigned stride0)
{
    unsigned long long ga = (unsigned long long)(uintptr_t)gptr;
    uint4v g0;
    g0[0] = 1u;                                         // count=1, user descriptor
    g0[1] = lds_off;                                    // lds_addr (bytes)
    g0[2] = (unsigned)(ga & 0xFFFFFFFFu);               // global_addr[31:0]
    g0[3] = (unsigned)((ga >> 32) & 0x01FFFFFFu)        // global_addr[56:32]
          | (2u << 30);                                 // type=2 ("image")
    uint8v g1;
    g1[0] = (1u << 16);                                 // wg_mask=0, data_size=1 (2B)
    g1[1] = (dim0 & 0xFFFFu) << 16;                     // tensor_dim0[15:0]
    g1[2] = (dim0 >> 16) | ((dim1 & 0xFFFFu) << 16);    // dim0[31:16] | dim1[15:0]
    g1[3] = (dim1 >> 16) | ((dim0 & 0xFFFFu) << 16);    // dim1[31:16] | tile_dim0
    g1[4] = (dim1 & 0xFFFFu);                           // tile_dim1 | tile_dim2=0
    g1[5] = stride0;                                    // tensor_dim0_stride[31:0]
    g1[6] = 0u;                                         // stride0[47:32] | stride1[15:0]
    g1[7] = 0u;                                         // stride1[47:16]
    uint4v gz = {0u, 0u, 0u, 0u};                       // groups 2/3 (2D tensor)
    asm volatile("tensor_load_to_lds %0, %1, %2, %3"
                 :: "s"(g0), "s"(g1), "s"(gz), "s"(gz)
                 : "memory");
}

// ---------------------------------------------------------------
// 1) maxed[r] = max over 32 support relations of cos_rel_all[sup, r]
// ---------------------------------------------------------------
__global__ __launch_bounds__(256) void k_support_max(
    const int* __restrict__ sup, const float* __restrict__ cosm,
    float* __restrict__ maxed)
{
    int r = blockIdx.x * 256 + threadIdx.x;
    if (r >= NUM_REL) return;
    float m = -3.0e38f;
    #pragma unroll 4
    for (int s = 0; s < S_; ++s)
        m = fmaxf(m, cosm[(size_t)sup[s] * NUM_REL + r]);
    maxed[r] = m;
}

// ---------------------------------------------------------------
// 2) f32 -> bf16 conversion (rel_emb, W_ih, W_hh)
// ---------------------------------------------------------------
__global__ __launch_bounds__(256) void k_cvt(
    const float* __restrict__ src, unsigned short* __restrict__ dst, int n)
{
    int i = blockIdx.x * 256 + threadIdx.x;
    if (i < n) dst[i] = bf16_of(src[i]);
}

// ---------------------------------------------------------------
// 3) frontier expansion + top-64 selection. One block per batch row.
// ---------------------------------------------------------------
template <int C>
__global__ __launch_bounds__(256) void k_expand(
    const int* __restrict__ cur,         // [B*C] entity ids
    const int* __restrict__ edge,        // [NUM_ENT, NEI, 2]
    const float* __restrict__ maxed,     // [NUM_REL]
    int* __restrict__ aim_ent, int* __restrict__ aim_rel, int* __restrict__ parent)
{
    const int b   = blockIdx.x;
    const int tid = threadIdx.x;
    const int tot = C * NEI;

    __shared__ float sc[C * NEI];
    __shared__ float rv[256];
    __shared__ int   ri[256];

    for (int i = tid; i < tot; i += 256) {
        int c = i >> 7, j = i & 127;
        int ent = cur[b * C + c];
        int rel = edge[((size_t)ent * NEI + j) * 2 + 1];
        sc[i] = maxed[rel];
    }
    __syncthreads();

    for (int t = 0; t < TOPK; ++t) {
        float best = -3.0e38f; int bi = tot;
        for (int i = tid; i < tot; i += 256) {
            float v = sc[i];
            if (v > best || (v == best && i < bi)) { best = v; bi = i; }
        }
        rv[tid] = best; ri[tid] = bi;
        __syncthreads();
        for (int off = 128; off; off >>= 1) {
            if (tid < off) {
                float ov = rv[tid + off]; int oi = ri[tid + off];
                if (ov > rv[tid] || (ov == rv[tid] && oi < ri[tid])) {
                    rv[tid] = ov; ri[tid] = oi;
                }
            }
            __syncthreads();
        }
        if (tid == 0) {
            int w = ri[0];
            sc[w] = -3.0e38f;
            int c = w >> 7, j = w & 127;
            int ent = cur[b * C + c];
            int ae  = edge[((size_t)ent * NEI + j) * 2 + 0];
            aim_ent[b * TOPK + t] = ae;
            aim_rel[b * TOPK + t] = edge[((size_t)ent * NEI + j) * 2 + 1];
            parent [b * TOPK + t] = c;
            // next level will gather this entity's edge row: prefetch it
            __builtin_prefetch(edge + (size_t)ae * NEI * 2, 0, 1);
        }
        __syncthreads();
    }
}

// ---------------------------------------------------------------
// 4) GRU cell via bf16 WMMA.  One wave = one 16-row M tile.
//    Weights (both bf16 matrices, 192 KB) are DMA'd into LDS once per
//    workgroup by the Tensor Data Mover; B fragments then come from LDS.
//    block = 256 threads = 8 waves = 128 rows; grid = 128 blocks.
// ---------------------------------------------------------------
__global__ __launch_bounds__(256) void k_gru(
    const int*  __restrict__ r_idx,      // [MROWS] relation per row
    const int*  __restrict__ p_idx,      // [MROWS] parent or nullptr
    const float* __restrict__ hprev,     // prev-step slice base or nullptr
    const unsigned short* __restrict__ relemb16,  // [NUM_REL, D] bf16
    const unsigned short* __restrict__ wih16,     // [NG, D] bf16
    const unsigned short* __restrict__ whh16,     // [NG, D] bf16
    const float* __restrict__ b_ih, const float* __restrict__ b_hh,
    float* __restrict__ hout)            // this-step slice base
{
    __shared__ unsigned short lds_wih[NG * D_];   // 96 KB
    __shared__ unsigned short lds_whh[NG * D_];   // 96 KB

    const int lane = threadIdx.x & 31;
    const int wv   = threadIdx.x >> 5;
    const int m16  = lane & 15;           // A row / B column / C column
    const int half = lane >> 4;
    const int rowbase = blockIdx.x * 128 + wv * 16;

    // ---- wave 0: TDM both weight matrices into LDS ----
    if (wv == 0) {
        tdm_load_2d_bf16(wih16, (unsigned)(uintptr_t)&lds_wih[0], D_, NG, D_);
        tdm_load_2d_bf16(whh16, (unsigned)(uintptr_t)&lds_whh[0], D_, NG, D_);
        __builtin_amdgcn_s_wait_tensorcnt(0);
    }

    // ---- meanwhile: per-lane A fragment gathers from global ----
    const int myrow = rowbase + m16;
    const int rel   = r_idx[myrow];
    const unsigned short* xrow = relemb16 + (size_t)rel * D_;
    v16bf ax[4];
    #pragma unroll
    for (int t = 0; t < 4; ++t) {
        const int o0 = t * 32 + half * 16;
        __builtin_memcpy(&ax[t], xrow + o0, 16);
        __builtin_memcpy((char*)&ax[t] + 16, xrow + o0 + 8, 16);
    }

    v16bf ah[4];
    if (hprev) {
        const int p  = p_idx[myrow];
        const int bb = myrow >> 6;
        const float* hrow = hprev + (size_t)bb * HB + (size_t)p * D_;
        #pragma unroll
        for (int t = 0; t < 4; ++t) {
            const int o0 = t * 32 + half * 16;
            v16bf a;
            #pragma unroll
            for (int i = 0; i < 8; ++i) {
                a[i]     = __builtin_bit_cast(__bf16, bf16_of(hrow[o0 + i]));
                a[8 + i] = __builtin_bit_cast(__bf16, bf16_of(hrow[o0 + 8 + i]));
            }
            ah[t] = a;
        }
    }

    __syncthreads();   // weights resident in LDS for all waves

    // ---- loop over 8 column tiles of the hidden dim ----
    #pragma unroll 1
    for (int dt = 0; dt < 8; ++dt) {
        v8f air = {0,0,0,0,0,0,0,0}, aiz = air, ain = air;
        v8f ahr = air, ahz = air, ahn = air;

        #pragma unroll
        for (int t = 0; t < 4; ++t) {
            const int ko = t * 32 + half * 16;   // contiguous 16 K values
            v16bf b0, b1, b2;
            __builtin_memcpy(&b0, &lds_wih[(size_t)(0 * D_ + dt * 16 + m16) * D_ + ko], 32);
            __builtin_memcpy(&b1, &lds_wih[(size_t)(1 * D_ + dt * 16 + m16) * D_ + ko], 32);
            __builtin_memcpy(&b2, &lds_wih[(size_t)(2 * D_ + dt * 16 + m16) * D_ + ko], 32);
            air = __builtin_amdgcn_wmma_f32_16x16x32_bf16(false, ax[t], false, b0, (short)0, air, false, false);
            aiz = __builtin_amdgcn_wmma_f32_16x16x32_bf16(false, ax[t], false, b1, (short)0, aiz, false, false);
            ain = __builtin_amdgcn_wmma_f32_16x16x32_bf16(false, ax[t], false, b2, (short)0, ain, false, false);
            if (hprev) {
                v16bf c0, c1, c2;
                __builtin_memcpy(&c0, &lds_whh[(size_t)(0 * D_ + dt * 16 + m16) * D_ + ko], 32);
                __builtin_memcpy(&c1, &lds_whh[(size_t)(1 * D_ + dt * 16 + m16) * D_ + ko], 32);
                __builtin_memcpy(&c2, &lds_whh[(size_t)(2 * D_ + dt * 16 + m16) * D_ + ko], 32);
                ahr = __builtin_amdgcn_wmma_f32_16x16x32_bf16(false, ah[t], false, c0, (short)0, ahr, false, false);
                ahz = __builtin_amdgcn_wmma_f32_16x16x32_bf16(false, ah[t], false, c1, (short)0, ahz, false, false);
                ahn = __builtin_amdgcn_wmma_f32_16x16x32_bf16(false, ah[t], false, c2, (short)0, ahn, false, false);
            }
        }

        // ---- gate epilogue ----
        const int col = dt * 16 + m16;
        const float bir = b_ih[0 * D_ + col], biz = b_ih[1 * D_ + col], bin = b_ih[2 * D_ + col];
        const float bhr = b_hh[0 * D_ + col], bhz = b_hh[1 * D_ + col], bhn = b_hh[2 * D_ + col];
        #pragma unroll
        for (int j = 0; j < 8; ++j) {
            const int mloc = j + 8 * half;          // C layout: VGPR j -> row j+8*half
            const int row  = rowbase + mloc;
            const int bb = row >> 6, kk = row & 63;
            float hv = 0.0f;
            if (hprev) {
                const int pp = p_idx[row];
                hv = hprev[(size_t)bb * HB + (size_t)pp * D_ + col];
            }
            const float rg = sigm(air[j] + bir + ahr[j] + bhr);
            const float zg = sigm(aiz[j] + biz + ahz[j] + bhz);
            const float ng = tanhf(ain[j] + bin + rg * (ahn[j] + bhn));
            hout[(size_t)bb * HB + (size_t)kk * D_ + col] = (1.0f - zg) * ng + zg * hv;
        }
    }
}

// ---------------------------------------------------------------
// 5) scalar outputs: tree_node, parent_index, parent_node, aim_rel
// ---------------------------------------------------------------
__global__ __launch_bounds__(256) void k_final(
    const int* __restrict__ qh,
    const int* __restrict__ e1, const int* __restrict__ r1,
    const int* __restrict__ e2, const int* __restrict__ r2, const int* __restrict__ p2,
    const int* __restrict__ e3, const int* __restrict__ r3, const int* __restrict__ p3,
    float* __restrict__ out)
{
    int idx = blockIdx.x * 256 + threadIdx.x;
    if (idx >= B_ * TOPK) return;
    const int b = idx / TOPK, k = idx % TOPK;
    float* tn = out + OFF_TN;
    float* pi = out + OFF_PI;
    float* pn = out + OFF_PN;
    float* ar = out + OFF_AR;
    const int base = b * 3 * TOPK + k;
    tn[base]            = (float)e1[idx];
    tn[base + TOPK]     = (float)e2[idx];
    tn[base + 2 * TOPK] = (float)e3[idx];
    pi[base]            = (float)p2[idx];
    pi[base + TOPK]     = (float)p3[idx];
    pi[base + 2 * TOPK] = (float)k;
    pn[base]            = (float)qh[b];
    pn[base + TOPK]     = (float)e1[b * TOPK + p2[idx]];
    pn[base + 2 * TOPK] = (float)e2[b * TOPK + p3[idx]];
    ar[base]            = (float)r1[b * TOPK + p2[idx]];
    ar[base + TOPK]     = (float)r2[b * TOPK + p3[idx]];
    ar[base + 2 * TOPK] = (float)r3[idx];
}

// ---------------------------------------------------------------
extern "C" void kernel_launch(void* const* d_in, const int* in_sizes, int n_in,
                              void* d_out, int out_size, void* d_ws, size_t ws_size,
                              hipStream_t stream)
{
    const int*   qh    = (const int*)  d_in[0];
    const int*   edge  = (const int*)  d_in[1];
    const int*   sup   = (const int*)  d_in[2];
    const float* cosm  = (const float*)d_in[3];
    const float* remb  = (const float*)d_in[4];
    const float* W_ih  = (const float*)d_in[5];
    const float* W_hh  = (const float*)d_in[6];
    const float* b_ih  = (const float*)d_in[7];
    const float* b_hh  = (const float*)d_in[8];
    float* out = (float*)d_out;

    char* ws = (char*)d_ws;
    float*          maxed    = (float*)(ws + WOFF_MAXED);
    unsigned short* relemb16 = (unsigned short*)(ws + WOFF_RELEMB);
    unsigned short* wih16    = (unsigned short*)(ws + WOFF_WIH);
    unsigned short* whh16    = (unsigned short*)(ws + WOFF_WHH);
    int* e1 = (int*)(ws + WOFF_INT + 0 * ISZ);
    int* r1 = (int*)(ws + WOFF_INT + 1 * ISZ);
    int* p1 = (int*)(ws + WOFF_INT + 2 * ISZ);
    int* e2 = (int*)(ws + WOFF_INT + 3 * ISZ);
    int* r2 = (int*)(ws + WOFF_INT + 4 * ISZ);
    int* p2 = (int*)(ws + WOFF_INT + 5 * ISZ);
    int* e3 = (int*)(ws + WOFF_INT + 6 * ISZ);
    int* r3 = (int*)(ws + WOFF_INT + 7 * ISZ);
    int* p3 = (int*)(ws + WOFF_INT + 8 * ISZ);

    float* hbase = out + OFF_TE;   // tree_emb_all [B, 3, K, D]

    k_support_max<<<(NUM_REL + 255) / 256, 256, 0, stream>>>(sup, cosm, maxed);
    k_cvt<<<(NUM_REL * D_ + 255) / 256, 256, 0, stream>>>(remb, relemb16, NUM_REL * D_);
    k_cvt<<<(NG * D_ + 255) / 256, 256, 0, stream>>>(W_ih, wih16, NG * D_);
    k_cvt<<<(NG * D_ + 255) / 256, 256, 0, stream>>>(W_hh, whh16, NG * D_);

    // level 1
    k_expand<1><<<B_, 256, 0, stream>>>(qh, edge, maxed, e1, r1, p1);
    k_gru<<<MROWS / 128, 256, 0, stream>>>(r1, nullptr, nullptr,
        relemb16, wih16, whh16, b_ih, b_hh, hbase + 0 * TOPK * D_);

    // level 2
    k_expand<TOPK><<<B_, 256, 0, stream>>>(e1, edge, maxed, e2, r2, p2);
    k_gru<<<MROWS / 128, 256, 0, stream>>>(r2, p2, hbase + 0 * TOPK * D_,
        relemb16, wih16, whh16, b_ih, b_hh, hbase + 1 * TOPK * D_);

    // level 3
    k_expand<TOPK><<<B_, 256, 0, stream>>>(e2, edge, maxed, e3, r3, p3);
    k_gru<<<MROWS / 128, 256, 0, stream>>>(r3, p3, hbase + 1 * TOPK * D_,
        relemb16, wih16, whh16, b_ih, b_hh, hbase + 2 * TOPK * D_);

    k_final<<<(B_ * TOPK + 255) / 256, 256, 0, stream>>>(
        qh, e1, r1, e2, r2, p2, e3, r3, p3, out);
}
